// Transformer_42606075576585
// MI455X (gfx1250) — matrix operand order repaired and checked by reading, not compiled, not run on hardware
//
#include <hip/hip_runtime.h>
#include <hip/hip_bf16.h>
#include <math.h>

typedef __attribute__((ext_vector_type(16))) _Float16 v16h;
typedef __attribute__((ext_vector_type(8)))  _Float16 h8;
typedef __attribute__((ext_vector_type(8)))  float    v8f;

#define B_   16
#define S_   512
#define E_   1024
#define H_   16
#define FF_  4096
#define L_   8
#define V_   100
#define DH_  64
#define BS_  (B_*S_)
#define EPS_ 1e-5f

// ---- CDNA5 async global->LDS staging (ASYNCcnt-tracked), with safe fallback ----
#if defined(__HIP_DEVICE_COMPILE__) && __has_builtin(__builtin_amdgcn_global_load_async_to_lds_b128)
#define USE_ASYNC_LDS 1
#else
#define USE_ASYNC_LDS 0
#endif

#if USE_ASYNC_LDS
// builtin expects: (int4 AS1* src, int4 AS3* dst, imm offset, imm cpol)
typedef int v4i_vs __attribute__((vector_size(16)));
typedef __attribute__((address_space(1))) v4i_vs glb_v4i;
typedef __attribute__((address_space(3))) v4i_vs lds_v4i;

static __device__ __forceinline__ void async_copy_b128(const _Float16* gsrc, _Float16* ldst) {
    __builtin_amdgcn_global_load_async_to_lds_b128(
        (glb_v4i*)gsrc,
        (lds_v4i*)ldst,
        0, 0);
}
static __device__ __forceinline__ void wait_async_lds() {
#if __has_builtin(__builtin_amdgcn_s_wait_asynccnt)
    __builtin_amdgcn_s_wait_asynccnt(0);
#else
    asm volatile("s_wait_asynccnt 0x0" ::: "memory");
#endif
}
#endif

// ---------------- WMMA GEMM: C[M,N] = A[M,K] * Bt[N,K]^T (+bias) ----------------
// A: f16 row-major (lda), Bt: f16 "B-transposed" [N][K] (ldb = K stride)
// Batched via blockIdx.z -> (b = z/hdiv, h = z%hdiv) with per-operand strides.
// GUARD=false: tile-exact dims (branch-free hot loop). GUARD=true: bounds-checked.
constexpr int BM = 128, BN = 64, BK = 32, LK = 48; // LK padded row (96B, 16B aligned)

template<bool WRITE_HALF, bool GUARD>
__global__ __launch_bounds__(256)
void gemm_f16_wmma(const _Float16* __restrict__ A, const _Float16* __restrict__ Bt,
                   void* __restrict__ Cv, const float* __restrict__ bias,
                   int M, int N, int K, int lda, int ldb, int ldc,
                   long long sAb, long long sAh, long long sBb, long long sBh,
                   long long sCb, long long sCh, int hdiv)
{
    __shared__ _Float16 As[BM * LK];
    __shared__ _Float16 Bs[BN * LK];

    const int tid   = threadIdx.x;
    const int lane  = tid & 31;
    const int wid   = tid >> 5;
    const int wm    = wid & 3;    // 4 waves along M (32 rows each)
    const int wn    = wid >> 2;   // 2 waves along N (32 cols each)
    const int laneN = lane & 15;
    const int half  = lane >> 4;  // K-half selector per ISA 16-bit layouts

    const int z  = blockIdx.z;
    const int bb = z / hdiv, hh = z % hdiv;
    const _Float16* Ab = A  + bb * sAb + hh * sAh;
    const _Float16* Bb = Bt + bb * sBb + hh * sBh;
    const long long coff = bb * sCb + hh * sCh;

    const int m0 = blockIdx.y * BM;
    const int n0 = blockIdx.x * BN;

    v8f c[2][2] = {};

    // staging roles: A = 128 rows x 2 half-rows(16 f16), B = 64 rows x 4 chunks(8 f16)
    const int arow = tid >> 1, ahalf  = tid & 1;   // A: 2 x b128 per thread
    const int brow = tid >> 2, bchunk = tid & 3;   // B: 1 x 8-f16 chunk per thread

    for (int k0 = 0; k0 < K; k0 += BK) {
#if USE_ASYNC_LDS
        if constexpr (!GUARD) {
            // direct global->LDS DMA (no VGPR round-trip); A: 1 x b128 per thread,
            // B: even-chunk threads move 16B (one 8-f16 chunk).
            const _Float16* asrc = Ab + (long long)(m0 + arow) * lda + k0 + ahalf * 16;
            async_copy_b128(asrc,     &As[arow * LK + ahalf * 16]);
            if ((bchunk & 1) == 0) {
                const _Float16* bsrc = Bb + (long long)(n0 + brow) * ldb + k0 + bchunk * 8;
                async_copy_b128(bsrc, &Bs[brow * LK + bchunk * 8]);
            }
            wait_async_lds();
        } else
#endif
        {
            // ---- stage A tile (guarded / fallback path) ----
            {
                const int gm = m0 + arow;
                h8 lo = {}, hi = {};
                if (!GUARD || gm < M) {
                    const _Float16* src = Ab + (long long)gm * lda + k0 + ahalf * 16;
                    lo = *(const h8*)(src);
                    hi = *(const h8*)(src + 8);
                }
                *(h8*)(&As[arow * LK + ahalf * 16])     = lo;
                *(h8*)(&As[arow * LK + ahalf * 16 + 8]) = hi;
            }
            // ---- stage B tile ----
            {
                const int gn = n0 + brow;
                h8 v = {};
                if (!GUARD || gn < N) v = *(const h8*)(Bb + (long long)gn * ldb + k0 + bchunk * 8);
                *(h8*)(&Bs[brow * LK + bchunk * 8]) = v;
            }
        }
        __syncthreads();

        // ---- fragments (ISA 7.12.2 16-bit layouts) ----
        v16h a[2], b[2];
#pragma unroll
        for (int fm = 0; fm < 2; ++fm) {
            const int rowA = wm * 32 + fm * 16 + laneN;
            h8 lo = *(const h8*)(&As[rowA * LK + half * 8]);       // K = half*8 .. +7
            h8 hi = *(const h8*)(&As[rowA * LK + 16 + half * 8]);  // K = 16+half*8 ..
            a[fm] = __builtin_shufflevector(lo, hi, 0,1,2,3,4,5,6,7,8,9,10,11,12,13,14,15);
        }
#pragma unroll
        for (int fn = 0; fn < 2; ++fn) {
            const int rowB = wn * 32 + fn * 16 + laneN;
            h8 lo = *(const h8*)(&Bs[rowB * LK + half * 8]);       // K = half*8 .. +7
            h8 hi = *(const h8*)(&Bs[rowB * LK + 16 + half * 8]);  // K = 16+half*8 ..
            b[fn] = __builtin_shufflevector(lo, hi, 0,1,2,3,4,5,6,7,8,9,10,11,12,13,14,15);
        }
#pragma unroll
        for (int fm = 0; fm < 2; ++fm)
#pragma unroll
            for (int fn = 0; fn < 2; ++fn)
                c[fm][fn] = __builtin_amdgcn_wmma_f32_16x16x32_f16(
                    false, a[fm], false, b[fn], (short)0, c[fm][fn], false, false);
        __syncthreads();
    }

    // ---- store (C layout: lanes 0-15 M=r, lanes 16-31 M=8+r; N=lane%16) ----
#pragma unroll
    for (int fm = 0; fm < 2; ++fm)
#pragma unroll
        for (int fn = 0; fn < 2; ++fn) {
            const int gn = n0 + wn * 32 + fn * 16 + laneN;
            const float bv = (bias && (!GUARD || gn < N)) ? bias[gn] : 0.0f;
#pragma unroll
            for (int r = 0; r < 8; ++r) {
                const int gm = m0 + wm * 32 + fm * 16 + half * 8 + r;
                if (!GUARD || (gm < M && gn < N)) {
                    const float v = c[fm][fn][r] + bv;
                    if constexpr (WRITE_HALF)
                        ((_Float16*)Cv)[coff + (long long)gm * ldc + gn] = (_Float16)v;
                    else
                        ((float*)Cv)[coff + (long long)gm * ldc + gn] = v;
                }
            }
        }
}

// ---------------- embedding + positional encoding + layernorm ----------------
__global__ __launch_bounds__(256)
void embed_ln_kernel(const int* __restrict__ tok, const float* __restrict__ emb,
                     const float* __restrict__ ns, const float* __restrict__ nb,
                     float* __restrict__ x)
{
    __shared__ float red[256];
    const int row = blockIdx.x;       // b*S + s
    const int s   = row % S_;
    const float* e = emb + (long long)tok[row] * E_;
    const float nldiv = -logf(10000.0f) / (float)E_;
    float v[4]; float sum = 0.f;
#pragma unroll
    for (int i = 0; i < 4; ++i) {
        const int c = threadIdx.x + i * 256;
        const float freq = expf((float)(2 * (c >> 1)) * nldiv);
        const float ang  = (float)s * freq;
        const float pe   = (c & 1) ? cosf(ang) : sinf(ang);
        v[i] = e[c] + pe;
        sum += v[i];
    }
    red[threadIdx.x] = sum; __syncthreads();
    for (int k = 128; k > 0; k >>= 1) { if (threadIdx.x < k) red[threadIdx.x] += red[threadIdx.x + k]; __syncthreads(); }
    const float mean = red[0] * (1.0f / E_); __syncthreads();
    float sq = 0.f;
#pragma unroll
    for (int i = 0; i < 4; ++i) { const float d = v[i] - mean; sq += d * d; }
    red[threadIdx.x] = sq; __syncthreads();
    for (int k = 128; k > 0; k >>= 1) { if (threadIdx.x < k) red[threadIdx.x] += red[threadIdx.x + k]; __syncthreads(); }
    const float rstd = rsqrtf(red[0] * (1.0f / E_) + EPS_);
#pragma unroll
    for (int i = 0; i < 4; ++i) {
        const int c = threadIdx.x + i * 256;
        x[(long long)row * E_ + c] = (v[i] - mean) * rstd * ns[c] + nb[c];
    }
}

// ---------------- layernorm: f32 in -> f16 out ----------------
__global__ __launch_bounds__(256)
void layernorm_kernel(const float* __restrict__ x, const float* __restrict__ sc,
                      const float* __restrict__ bi, _Float16* __restrict__ out16)
{
    __shared__ float red[256];
    const long long row = blockIdx.x;
    const float* src = x + row * E_;
    float v[4]; float sum = 0.f;
#pragma unroll
    for (int i = 0; i < 4; ++i) { v[i] = src[threadIdx.x + i * 256]; sum += v[i]; }
    red[threadIdx.x] = sum; __syncthreads();
    for (int k = 128; k > 0; k >>= 1) { if (threadIdx.x < k) red[threadIdx.x] += red[threadIdx.x + k]; __syncthreads(); }
    const float mean = red[0] * (1.0f / E_); __syncthreads();
    float sq = 0.f;
#pragma unroll
    for (int i = 0; i < 4; ++i) { const float d = v[i] - mean; sq += d * d; }
    red[threadIdx.x] = sq; __syncthreads();
    for (int k = 128; k > 0; k >>= 1) { if (threadIdx.x < k) red[threadIdx.x] += red[threadIdx.x + k]; __syncthreads(); }
    const float rstd = rsqrtf(red[0] * (1.0f / E_) + EPS_);
#pragma unroll
    for (int i = 0; i < 4; ++i) {
        const int c = threadIdx.x + i * 256;
        out16[row * E_ + c] = (_Float16)((v[i] - mean) * rstd * sc[c] + bi[c]);
    }
}

// ---------------- masked, scaled softmax: f32 scores -> f16 probs ----------------
__global__ __launch_bounds__(256)
void softmax_mask_kernel(const float* __restrict__ scores, const int* __restrict__ tokens,
                         _Float16* __restrict__ probs)
{
    __shared__ float red[256];
    const int row = blockIdx.x;              // b*H*S + h*S + s
    const int b   = row / (H_ * S_);
    const float scale = 0.125f;              // 1/sqrt(Dh)
    const float* src = scores + (long long)row * S_;
    const int t0 = threadIdx.x;
    const float v0 = (tokens[b * S_ + t0]       != 0) ? src[t0]       * scale : -1e9f;
    const float v1 = (tokens[b * S_ + t0 + 256] != 0) ? src[t0 + 256] * scale : -1e9f;
    red[t0] = fmaxf(v0, v1); __syncthreads();
    for (int k = 128; k > 0; k >>= 1) { if (t0 < k) red[t0] = fmaxf(red[t0], red[t0 + k]); __syncthreads(); }
    const float m = red[0]; __syncthreads();
    const float e0 = expf(v0 - m), e1 = expf(v1 - m);
    red[t0] = e0 + e1; __syncthreads();
    for (int k = 128; k > 0; k >>= 1) { if (t0 < k) red[t0] += red[t0 + k]; __syncthreads(); }
    const float inv = 1.0f / red[0];
    _Float16* dst = probs + (long long)row * S_;
    dst[t0]       = (_Float16)(e0 * inv);
    dst[t0 + 256] = (_Float16)(e1 * inv);
}

// ---------------- elementwise helpers ----------------
__global__ void residual_add_kernel(float* __restrict__ x, const float* __restrict__ c, long long n) {
    for (long long i = (long long)blockIdx.x * blockDim.x + threadIdx.x; i < n;
         i += (long long)gridDim.x * blockDim.x) x[i] += c[i];
}
__global__ void relu_cast_kernel(const float* __restrict__ c, _Float16* __restrict__ o, long long n) {
    for (long long i = (long long)blockIdx.x * blockDim.x + threadIdx.x; i < n;
         i += (long long)gridDim.x * blockDim.x) o[i] = (_Float16)fmaxf(c[i], 0.0f);
}
__global__ void cast_f16_kernel(const float* __restrict__ x, _Float16* __restrict__ o, long long n) {
    for (long long i = (long long)blockIdx.x * blockDim.x + threadIdx.x; i < n;
         i += (long long)gridDim.x * blockDim.x) o[i] = (_Float16)x[i];
}
// W [H,E,Dh] f32 -> Bt [H*Dh][E] f16
__global__ void repack_qkv_kernel(const float* __restrict__ w, _Float16* __restrict__ out) {
    const long long total = (long long)H_ * E_ * DH_;
    for (long long o = (long long)blockIdx.x * blockDim.x + threadIdx.x; o < total;
         o += (long long)gridDim.x * blockDim.x) {
        const int n = (int)(o / E_), e = (int)(o % E_);
        const int h = n / DH_, d = n % DH_;
        out[o] = (_Float16)w[((long long)h * E_ + e) * DH_ + d];
    }
}
// W [K][N] f32 -> Bt [N][K] f16
__global__ void repack_t_kernel(const float* __restrict__ w, _Float16* __restrict__ out, int K, int N) {
    const long long total = (long long)K * N;
    for (long long o = (long long)blockIdx.x * blockDim.x + threadIdx.x; o < total;
         o += (long long)gridDim.x * blockDim.x) {
        const int n = (int)(o / K), k = (int)(o % K);
        out[o] = (_Float16)w[(long long)k * N + n];
    }
}
// v [B,S,H*Dh] f16 -> vt [B,H,Dh,S] f16
__global__ void transpose_v_kernel(const _Float16* __restrict__ v, _Float16* __restrict__ vt) {
    const long long total = (long long)B_ * H_ * DH_ * S_;
    for (long long o = (long long)blockIdx.x * blockDim.x + threadIdx.x; o < total;
         o += (long long)gridDim.x * blockDim.x) {
        const int t = (int)(o % S_); long long r = o / S_;
        const int d = (int)(r % DH_); r /= DH_;
        const int h = (int)(r % H_);  const int b = (int)(r / H_);
        vt[o] = v[((long long)b * S_ + t) * E_ + h * DH_ + d];
    }
}
__global__ void zero_scalar_kernel(float* p) { if (threadIdx.x == 0 && blockIdx.x == 0) *p = 0.0f; }

// ---------------- cross-entropy: mean NLL over all rows ----------------
__global__ __launch_bounds__(128)
void ce_loss_kernel(const float* __restrict__ logits, const int* __restrict__ target,
                    float* __restrict__ loss)
{
    __shared__ float red[128];
    const int row = blockIdx.x;
    const float* src = logits + (long long)row * V_;
    const int t = threadIdx.x;
    const float v = (t < V_) ? src[t] : -3.4e38f;
    red[t] = v; __syncthreads();
    for (int k = 64; k > 0; k >>= 1) { if (t < k) red[t] = fmaxf(red[t], red[t + k]); __syncthreads(); }
    const float m = red[0]; __syncthreads();
    red[t] = (t < V_) ? expf(v - m) : 0.0f; __syncthreads();
    for (int k = 64; k > 0; k >>= 1) { if (t < k) red[t] += red[t + k]; __syncthreads(); }
    if (t == 0) {
        const float lse = m + logf(red[0]);
        atomicAdd(loss, (lse - src[target[row]]) * (1.0f / (float)(B_ * S_)));
    }
}

// ---------------- host orchestration ----------------
extern "C" void kernel_launch(void* const* d_in, const int* in_sizes, int n_in,
                              void* d_out, int out_size, void* d_ws, size_t ws_size,
                              hipStream_t stream)
{
    (void)in_sizes; (void)n_in; (void)out_size; (void)ws_size;
    const int*   inputs = (const int*)  d_in[0];
    const int*   target = (const int*)  d_in[1];
    const float* emb    = (const float*)d_in[2];
    const float* Wq     = (const float*)d_in[3];
    const float* Wk     = (const float*)d_in[4];
    const float* Wv     = (const float*)d_in[5];
    const float* bq     = (const float*)d_in[6];
    const float* bk     = (const float*)d_in[7];
    const float* bv     = (const float*)d_in[8];
    const float* Wo     = (const float*)d_in[9];
    const float* bo     = (const float*)d_in[10];
    const float* ln1_s  = (const float*)d_in[11];
    const float* ln1_b  = (const float*)d_in[12];
    const float* ln2_s  = (const float*)d_in[13];
    const float* ln2_b  = (const float*)d_in[14];
    const float* W1     = (const float*)d_in[15];
    const float* b1     = (const float*)d_in[16];
    const float* W2     = (const float*)d_in[17];
    const float* b2     = (const float*)d_in[18];
    const float* norm_s = (const float*)d_in[19];
    const float* norm_b = (const float*)d_in[20];
    const float* out_W  = (const float*)d_in[21];
    const float* out_b  = (const float*)d_in[22];

    float* logits = (float*)d_out;
    float* loss   = logits + (long long)BS_ * V_;

    // ---- carve workspace ----
    unsigned char* p = (unsigned char*)d_ws;
    auto carve = [&](size_t bytes) -> void* {
        void* r = (void*)p; p += (bytes + 255) & ~(size_t)255; return r;
    };
    float*    x      = (float*)   carve((size_t)BS_ * E_ * 4);
    _Float16* xh     = (_Float16*)carve((size_t)BS_ * E_ * 2);
    _Float16* qh     = (_Float16*)carve((size_t)BS_ * E_ * 2);
    _Float16* kh     = (_Float16*)carve((size_t)BS_ * E_ * 2);
    _Float16* vh     = (_Float16*)carve((size_t)BS_ * E_ * 2);
    _Float16* vt     = (_Float16*)carve((size_t)BS_ * E_ * 2);
    _Float16* oh     = (_Float16*)carve((size_t)BS_ * E_ * 2);
    float*    scores = (float*)   carve((size_t)B_ * H_ * S_ * S_ * 4); // also generic f32 GEMM scratch
    _Float16* probs  = (_Float16*)carve((size_t)B_ * H_ * S_ * S_ * 2);
    _Float16* ffh    = (_Float16*)carve((size_t)BS_ * FF_ * 2);
    _Float16* wqT    = (_Float16*)carve((size_t)E_ * E_ * 2);
    _Float16* wkT    = (_Float16*)carve((size_t)E_ * E_ * 2);
    _Float16* wvT    = (_Float16*)carve((size_t)E_ * E_ * 2);
    _Float16* woT    = (_Float16*)carve((size_t)E_ * E_ * 2);
    _Float16* w1T    = (_Float16*)carve((size_t)E_ * FF_ * 2);
    _Float16* w2T    = (_Float16*)carve((size_t)FF_ * E_ * 2);
    _Float16* owT    = (_Float16*)carve((size_t)E_ * V_ * 2);
    float*    c32    = scores;  // alias: max f32 GEMM output (BS*FF*4) < scores size

    auto gemm = [&](const _Float16* A, const _Float16* Bt, void* C, const float* bias,
                    int M, int N, int K, int lda, int ldb, int ldc,
                    long long sAb, long long sAh, long long sBb, long long sBh,
                    long long sCb, long long sCh, int hdiv, int batches, bool half, bool guard) {
        dim3 g((N + BN - 1) / BN, (M + BM - 1) / BM, batches);
        if (guard)
            gemm_f16_wmma<false, true><<<g, 256, 0, stream>>>(A, Bt, C, bias, M, N, K, lda, ldb, ldc,
                                                              sAb, sAh, sBb, sBh, sCb, sCh, hdiv);
        else if (half)
            gemm_f16_wmma<true, false><<<g, 256, 0, stream>>>(A, Bt, C, bias, M, N, K, lda, ldb, ldc,
                                                              sAb, sAh, sBb, sBh, sCb, sCh, hdiv);
        else
            gemm_f16_wmma<false, false><<<g, 256, 0, stream>>>(A, Bt, C, bias, M, N, K, lda, ldb, ldc,
                                                               sAb, sAh, sBb, sBh, sCb, sCh, hdiv);
    };
    auto ew = [](long long n) { long long b = (n + 255) / 256; return (int)(b < 16384 ? b : 16384); };

    const long long nBE  = (long long)BS_ * E_;
    const long long nBFF = (long long)BS_ * FF_;

    // embedding + PE + initial layernorm -> x (f32 residual stream)
    embed_ln_kernel<<<BS_, 256, 0, stream>>>(inputs, emb, norm_s, norm_b, x);

    for (int l = 0; l < L_; ++l) {
        const long long oW = (long long)l * H_ * E_ * DH_;
        // weight repacks (f32 -> f16 [N][K])
        repack_qkv_kernel<<<4096, 256, 0, stream>>>(Wq + oW, wqT);
        repack_qkv_kernel<<<4096, 256, 0, stream>>>(Wk + oW, wkT);
        repack_qkv_kernel<<<4096, 256, 0, stream>>>(Wv + oW, wvT);
        repack_t_kernel  <<<4096, 256, 0, stream>>>(Wo + (long long)l * E_ * E_,  woT, E_, E_);
        repack_t_kernel  <<<8192, 256, 0, stream>>>(W1 + (long long)l * E_ * FF_, w1T, E_, FF_);
        repack_t_kernel  <<<8192, 256, 0, stream>>>(W2 + (long long)l * FF_ * E_, w2T, FF_, E_);

        // ln1 -> xh (f16)
        layernorm_kernel<<<BS_, 256, 0, stream>>>(x, ln1_s + l * E_, ln1_b + l * E_, xh);

        // q,k,v = xn @ W + b   (f16 outputs)
        gemm(xh, wqT, qh, bq + l * E_, BS_, E_, E_, E_, E_, E_, 0,0,0,0,0,0, 1, 1, true,  false);
        gemm(xh, wkT, kh, bk + l * E_, BS_, E_, E_, E_, E_, E_, 0,0,0,0,0,0, 1, 1, true,  false);
        gemm(xh, wvT, vh, bv + l * E_, BS_, E_, E_, E_, E_, E_, 0,0,0,0,0,0, 1, 1, true,  false);
        transpose_v_kernel<<<8192, 256, 0, stream>>>(vh, vt);

        // scores[b,h] = q[b,h] @ k[b,h]^T   (Bt layout of k^T is just k row-major)
        gemm(qh, kh, scores, nullptr, S_, S_, DH_, E_, E_, S_,
             (long long)S_ * E_, DH_, (long long)S_ * E_, DH_,
             (long long)H_ * S_ * S_, (long long)S_ * S_, H_, B_ * H_, false, false);

        softmax_mask_kernel<<<B_ * H_ * S_, 256, 0, stream>>>(scores, inputs, probs);

        // o[b,h] = probs[b,h] @ v[b,h]  -> head-concat layout [B,S,E] f16
        gemm(probs, vt, oh, nullptr, S_, DH_, S_, S_, S_, E_,
             (long long)H_ * S_ * S_, (long long)S_ * S_,
             (long long)H_ * DH_ * S_, (long long)DH_ * S_,
             (long long)S_ * E_, DH_, H_, B_ * H_, true, false);

        // x += o @ Wo + bo
        gemm(oh, woT, c32, bo + l * E_, BS_, E_, E_, E_, E_, E_, 0,0,0,0,0,0, 1, 1, false, false);
        residual_add_kernel<<<ew(nBE), 256, 0, stream>>>(x, c32, nBE);

        // ln2 -> xh; ff = relu(xn2 @ W1 + b1) @ W2 + b2; x += ff
        layernorm_kernel<<<BS_, 256, 0, stream>>>(x, ln2_s + l * E_, ln2_b + l * E_, xh);
        gemm(xh, w1T, c32, b1 + (long long)l * FF_, BS_, FF_, E_, E_, E_, FF_, 0,0,0,0,0,0, 1, 1, false, false);
        relu_cast_kernel<<<ew(nBFF), 256, 0, stream>>>(c32, ffh, nBFF);
        gemm(ffh, w2T, c32, b2 + l * E_, BS_, E_, FF_, FF_, FF_, E_, 0,0,0,0,0,0, 1, 1, false, false);
        residual_add_kernel<<<ew(nBE), 256, 0, stream>>>(x, c32, nBE);
    }

    // logits = x @ out_W + out_b  (N=100 -> guarded GEMM instantiation)
    repack_t_kernel<<<1024, 256, 0, stream>>>(out_W, owT, E_, V_);
    cast_f16_kernel<<<ew(nBE), 256, 0, stream>>>(x, xh, nBE);
    gemm(xh, owT, logits, out_b, BS_, V_, E_, E_, E_, V_, 0,0,0,0,0,0, 1, 1, false, true);

    // mean cross-entropy
    zero_scalar_kernel<<<1, 32, 0, stream>>>(loss);
    ce_loss_kernel<<<BS_, 128, 0, stream>>>(logits, target, loss);
}
